// FrozenHopfield_33612414058958
// MI455X (gfx1250) — matrix operand order repaired and checked by reading, not compiled, not run on hardware
//
#include <hip/hip_runtime.h>
#include <stdint.h>

// ---------------------------------------------------------------------------
// FrozenHopfield on MI455X (gfx1250, wave32, WMMA + async LDS copies)
//
// Hot op: cosine-sims GEMM, 2*2048*50257*768 = 158 GFLOP in bf16 WMMA.
// embeddings bf16 (77MB) are L2-resident (192MB L2). 128x128 workgroup tiles
// keep B-operand L2 traffic at 16x77MB ~= 1.2GB. K is streamed in 32-wide
// slabs staged via GLOBAL_LOAD_ASYNC_TO_LDS_B128 (ASYNCcnt, double-buffered)
// so L2 latency overlaps WMMA. Each wave computes 16 rows x 128 cols: one A
// fragment reused across 8 WMMAs per K-step; top-32 upkeep is wave-local.
// sims[2048,50257] (412MB) is never materialized in HBM.
// ---------------------------------------------------------------------------

typedef float    v8f   __attribute__((ext_vector_type(8)));
typedef __bf16   v16bf __attribute__((ext_vector_type(16)));
typedef uint32_t u32x4 __attribute__((ext_vector_type(4)));

#define BT      2048          // B*T rows
#define DIN     512
#define DE      768
#define NV      50257
#define VPAD    51200         // multiple of NCHUNK*128
#define NCHUNK  16
#define VCHUNK  (VPAD / NCHUNK)   // 3200 = 25 * 128
#define TOPK    32
#define MTILE   128           // rows per workgroup in sims kernel
#define FH_BETA 10.0f
#define FH_EPS  1e-8f

// dynamic LDS layout for sims_topk
#define SA_STRIDE 40              // 32 + 8 pad: conflict-free ds_read_b128
#define SLAB      (MTILE * SA_STRIDE)                     // elements per slab
#define STG_STRIDE 136            // 128 + 8 pad
#define SMEM_SLABS (4 * SLAB * 2)                         // A0,A1,B0,B1 = 40960
#define SMEM_STG   (MTILE * STG_STRIDE * 2)               // 34816
#define SMEM_HN    (MTILE * 4)                            // 512
#define SMEM_KV    (MTILE * TOPK * 4)                     // 16384
#define SMEM_KI    (MTILE * TOPK * 4)                     // 16384
#define SMEM_TOTAL (SMEM_SLABS + SMEM_STG + SMEM_HN + SMEM_KV + SMEM_KI) // 109056

__device__ __forceinline__ unsigned short f2bf(float f) {
  union { float f; uint32_t u; } x; x.f = f;
  uint32_t u = x.u;
  u += 0x7FFFu + ((u >> 16) & 1u);   // round-to-nearest-even
  return (unsigned short)(u >> 16);
}
__device__ __forceinline__ float bf2f(unsigned short b) {
  union { float f; uint32_t u; } x; x.u = ((uint32_t)b) << 16;
  return x.f;
}

// CDNA5 async memory->LDS copy, tracked by ASYNCcnt (in-order completion).
// lds_off = wave-relative LDS byte offset (low 32 bits of generic pointer).
__device__ __forceinline__ void async_to_lds_b128(uint32_t lds_off, const void* gptr) {
  asm volatile("global_load_async_to_lds_b128 %0, %1, off"
               :: "v"(lds_off), "v"(gptr) : "memory");
}
__device__ __forceinline__ uint32_t lds_off32(const void* p) {
  return (uint32_t)(size_t)p;
}

union FragAB {                 // one WMMA A/B operand: 16 bf16 per lane
  unsigned short u[16];
  u32x4          q[2];
  v16bf          v;
};

// ---------------------------------------------------------------------------
// Kernel 1: embeddings -> bf16 (padded) + per-row L2 norm
// ---------------------------------------------------------------------------
__global__ void prep_emb(const float* __restrict__ emb,
                         unsigned short* __restrict__ emb_bf,
                         float* __restrict__ e_norm) {
  const int row = blockIdx.x;       // 0..VPAD-1
  const int t   = threadIdx.x;      // 256
  __shared__ float red[256];
  float ss = 0.f;
  if (row < NV) {
    for (int d = t; d < DE; d += 256) {
      float f = emb[(size_t)row * DE + d];
      emb_bf[(size_t)row * DE + d] = f2bf(f);
      ss += f * f;
    }
  } else {
    for (int d = t; d < DE; d += 256) emb_bf[(size_t)row * DE + d] = 0;
  }
  red[t] = ss; __syncthreads();
  for (int s = 128; s > 0; s >>= 1) {
    if (t < s) red[t] += red[t + s];
    __syncthreads();
  }
  if (t == 0) e_norm[row] = (row < NV) ? sqrtf(red[0]) : 1.0f;
}

// ---------------------------------------------------------------------------
// Kernel 2: h = x @ proj^T  (M=2048,N=768,K=512) via bf16 WMMA.
// ---------------------------------------------------------------------------
__global__ void __launch_bounds__(128)
proj_kernel(const float* __restrict__ x, const float* __restrict__ proj,
            unsigned short* __restrict__ h_bf, float* __restrict__ h_norm) {
  const int tile = blockIdx.x;            // 0..127
  const int t    = threadIdx.x;           // 128 = 4 waves
  const int lane = t & 31, wave = t >> 5;
  const int hi   = lane >> 4, mrow = lane & 15;

  __shared__ unsigned short sh_x[16 * 520];   // 512 + 8 pad
  __shared__ float hn2[16];

  for (int i = t; i < 16 * DIN; i += 128) {
    int m = i / DIN, k = i % DIN;
    sh_x[m * 520 + k] = f2bf(x[(size_t)(tile * 16 + m) * DIN + k]);
  }
  if (t < 16) hn2[t] = 0.f;
  __syncthreads();

  for (int nt = wave; nt < DE / 16; nt += 4) {
    const int nb   = nt * 16;
    const int ncol = nb + (lane & 15);
    v8f c = {};
    for (int kb = 0; kb < DIN; kb += 32) {
      FragAB a, b;
      a.q[0] = *(const u32x4*)&sh_x[mrow * 520 + kb +      8 * hi];
      a.q[1] = *(const u32x4*)&sh_x[mrow * 520 + kb + 16 + 8 * hi];
      const float* pb = &proj[(size_t)ncol * DIN + kb + 8 * hi];
      #pragma unroll
      for (int j = 0; j < 16; ++j) b.u[j] = f2bf(pb[j < 8 ? j : j + 8]);
      c = __builtin_amdgcn_wmma_f32_16x16x32_bf16(false, a.v, false, b.v,
                                                  (short)0, c, false, false);
    }
    #pragma unroll
    for (int r = 0; r < 8; ++r) {
      int   m   = r + 8 * hi;
      int   row = tile * 16 + m;
      float v   = c[r];
      h_bf[(size_t)row * DE + ncol] = f2bf(v);
      atomicAdd(&hn2[m], v * v);            // ds_add_f32
    }
  }
  __syncthreads();
  if (t < 16) h_norm[tile * 16 + t] = sqrtf(hn2[t]);
}

// ---------------------------------------------------------------------------
// Kernel 3 (hot): fused cosine-sims GEMM (128x128 tiles) + per-chunk top-32.
// K streamed in 32-wide slabs, double-buffered via async LDS copies.
// ---------------------------------------------------------------------------
__global__ void __launch_bounds__(256)
sims_topk(const unsigned short* __restrict__ emb_bf,
          const float* __restrict__ e_norm,
          const unsigned short* __restrict__ h_bf,
          const float* __restrict__ h_norm,
          float* __restrict__ pvals, int* __restrict__ pidx) {
  const int row0  = blockIdx.x * MTILE;   // 128-row M tile
  const int chunk = blockIdx.y;           // V chunk
  const int t     = threadIdx.x;          // 256 = 8 waves
  const int lane  = t & 31, wave = t >> 5;
  const int hi    = lane >> 4, nlane = lane & 15;

  extern __shared__ char smem[];
  unsigned short* sAb = (unsigned short*)smem;            // [2][128][40]
  unsigned short* sBb = sAb + 2 * SLAB;                   // [2][128][40]
  unsigned short* stg = sBb + 2 * SLAB;                   // [128][136] bf16 sims
  float* hn = (float*)(stg + MTILE * STG_STRIDE);         // [128]
  float* kv = hn + MTILE;                                 // [128][32]
  int*   ki = (int*)(kv + MTILE * TOPK);                  // [128][32]

  if (t < MTILE) hn[t] = h_norm[row0 + t];
  for (int i = t; i < MTILE * TOPK; i += 256) {
    kv[i] = -3.0e38f; ki[i] = 0;
  }

  // cooperative copy slot: thread t stages row (t>>1), k-half (t&1) (one b128 = 8 bf16)
  const int cm = t >> 1, chalf = (t & 1) * 16;
  const uint32_t ldsA0 = lds_off32(&sAb[cm * SA_STRIDE + chalf]);
  const uint32_t ldsB0 = lds_off32(&sBb[cm * SA_STRIDE + chalf]);
  const unsigned short* gA = &h_bf[(size_t)(row0 + cm) * DE + chalf];

  const int col0 = chunk * VCHUNK;
  for (int it = 0; it < VCHUNK / MTILE; ++it) {
    const int cbase = col0 + it * MTILE;
    const unsigned short* gB = &emb_bf[(size_t)(cbase + cm) * DE + chalf];

    v8f acc[8];
    #pragma unroll
    for (int nf = 0; nf < 8; ++nf) acc[nf] = (v8f){};

    // prologue: slab 0 -> buffer 0 (ASYNCcnt = 2 per wave)
    async_to_lds_b128(ldsA0, gA);
    async_to_lds_b128(ldsB0, gB);

    for (int ks = 0; ks < DE / 32; ++ks) {
      const int buf  = ks & 1;
      const bool more = (ks + 1) < DE / 32;
      if (more) {       // stage next slab into the other buffer (cnt -> 4)
        const int kb2 = (ks + 1) * 32;
        const uint32_t nboff = (uint32_t)(buf ^ 1) * (SLAB * 2);
        async_to_lds_b128(ldsA0 + nboff, gA + kb2);
        async_to_lds_b128(ldsB0 + nboff, gB + kb2);
        asm volatile("s_wait_asynccnt 0x2" ::: "memory"); // in-order: cur buf done
      } else {
        asm volatile("s_wait_asynccnt 0x0" ::: "memory");
      }
      __syncthreads();

      FragAB a;
      a.q[0] = *(const u32x4*)&sAb[buf * SLAB + (wave * 16 + nlane) * SA_STRIDE +      8 * hi];
      a.q[1] = *(const u32x4*)&sAb[buf * SLAB + (wave * 16 + nlane) * SA_STRIDE + 16 + 8 * hi];
      #pragma unroll
      for (int nf = 0; nf < 8; ++nf) {
        FragAB b;
        const unsigned short* bp = &sBb[buf * SLAB + (nf * 16 + nlane) * SA_STRIDE + 8 * hi];
        b.q[0] = *(const u32x4*)bp;
        b.q[1] = *(const u32x4*)(bp + 16);
        acc[nf] = __builtin_amdgcn_wmma_f32_16x16x32_bf16(false, a.v, false, b.v,
                                                          (short)0, acc[nf], false, false);
      }
      __syncthreads();   // all reads of cur buf done before it is refilled
    }

    // scale to cosine and stage (wave-private rows)
    #pragma unroll
    for (int nf = 0; nf < 8; ++nf) {
      const int col = cbase + nf * 16 + nlane;
      const float en = e_norm[col];
      #pragma unroll
      for (int r = 0; r < 8; ++r) {
        const int ml = wave * 16 + r + 8 * hi;
        float s = acc[nf][r] / (hn[ml] * en + FH_EPS);
        stg[ml * STG_STRIDE + nf * 16 + nlane] = (col < NV) ? f2bf(s) : (unsigned short)0xFF7F;
      }
    }
    __syncthreads();

    // wave-local top-32 upkeep: lanes 0..15 each own one row of this wave
    if (lane < 16) {
      const int ml = wave * 16 + lane;
      float* rv = &kv[ml * TOPK]; int* ri = &ki[ml * TOPK];
      float mn = rv[0]; int mp = 0;
      #pragma unroll
      for (int q = 1; q < TOPK; ++q) if (rv[q] < mn) { mn = rv[q]; mp = q; }
      for (int j = 0; j < MTILE; ++j) {
        float v = bf2f(stg[ml * STG_STRIDE + j]);
        if (v > mn) {
          rv[mp] = v; ri[mp] = cbase + j;
          mn = rv[0]; mp = 0;
          for (int q = 1; q < TOPK; ++q) if (rv[q] < mn) { mn = rv[q]; mp = q; }
        }
      }
    }
    __syncthreads();
  }

  for (int i = t; i < MTILE * TOPK; i += 256) {
    int m = i >> 5, k = i & 31;
    size_t o = ((size_t)(row0 + m) * NCHUNK + chunk) * TOPK + k;
    pvals[o] = kv[i];
    pidx[o]  = ki[i];
  }
}

// ---------------------------------------------------------------------------
// Kernel 4: per row merge 16x32 partials -> top-32, softmax(beta*v),
// weighted sum of fp32 embedding rows (L2-resident gathers).
// ---------------------------------------------------------------------------
__global__ void __launch_bounds__(256)
merge_out(const float* __restrict__ pvals, const int* __restrict__ pidx,
          const float* __restrict__ emb, float* __restrict__ out) {
  const int row = blockIdx.x;        // 0..2047
  const int t   = threadIdx.x;       // 256
  __shared__ float cv[512]; __shared__ int ci[512];
  __shared__ float rv[256]; __shared__ int rp[256];
  __shared__ float sv[TOPK]; __shared__ int si[TOPK];
  __shared__ float w[TOPK];

  for (int i = t; i < 512; i += 256) {
    cv[i] = pvals[(size_t)row * (NCHUNK * TOPK) + i];
    ci[i] = pidx [(size_t)row * (NCHUNK * TOPK) + i];
  }
  __syncthreads();

  for (int k = 0; k < TOPK; ++k) {   // iterative parallel argmax
    float bv = cv[t]; int bp = t;
    if (cv[t + 256] > bv) { bv = cv[t + 256]; bp = t + 256; }
    rv[t] = bv; rp[t] = bp;
    __syncthreads();
    for (int s = 128; s > 0; s >>= 1) {
      if (t < s && rv[t + s] > rv[t]) { rv[t] = rv[t + s]; rp[t] = rp[t + s]; }
      __syncthreads();
    }
    if (t == 0) {
      int p = rp[0];
      sv[k] = cv[p]; si[k] = ci[p];
      cv[p] = -3.0e38f;
    }
    __syncthreads();
  }

  if (t == 0) {                      // softmax over 32 (sv[0] is the max)
    float mx = sv[0], sum = 0.f;
    for (int k = 0; k < TOPK; ++k) { float e = __expf(FH_BETA * (sv[k] - mx)); w[k] = e; sum += e; }
    float inv = 1.f / sum;
    for (int k = 0; k < TOPK; ++k) w[k] *= inv;
  }
  __syncthreads();

  for (int d = t; d < DE; d += 256) {
    float acc = 0.f;
    #pragma unroll 8
    for (int k = 0; k < TOPK; ++k) acc += w[k] * emb[(size_t)si[k] * DE + d];
    out[(size_t)row * DE + d] = acc;
  }
}

// ---------------------------------------------------------------------------
extern "C" void kernel_launch(void* const* d_in, const int* in_sizes, int n_in,
                              void* d_out, int out_size, void* d_ws, size_t ws_size,
                              hipStream_t stream) {
  const float* x    = (const float*)d_in[0];   // [2,1024,512]
  const float* emb  = (const float*)d_in[1];   // [50257,768]
  const float* proj = (const float*)d_in[2];   // [768,512]
  float* out = (float*)d_out;                  // [2,1024,768]

  // workspace layout (~90.4 MB, all 256B-aligned offsets)
  char* ws = (char*)d_ws;
  size_t off = 0;
  unsigned short* emb_bf = (unsigned short*)(ws + off); off += (size_t)VPAD * DE * 2;
  float*          e_norm = (float*)(ws + off);          off += (size_t)VPAD * 4;
  unsigned short* h_bf   = (unsigned short*)(ws + off); off += (size_t)BT * DE * 2;
  float*          h_norm = (float*)(ws + off);          off += (size_t)BT * 4;
  float*          pvals  = (float*)(ws + off);          off += (size_t)BT * NCHUNK * TOPK * 4;
  int*            pidx   = (int*)(ws + off);

  // opt in to >64KB dynamic LDS for the hot kernel (WGP has 320KB)
  static int smem_attr_set = 0;
  if (!smem_attr_set) {
    hipFuncSetAttribute((const void*)sims_topk,
                        hipFuncAttributeMaxDynamicSharedMemorySize, SMEM_TOTAL);
    smem_attr_set = 1;
  }

  prep_emb  <<<VPAD, 256, 0, stream>>>(emb, emb_bf, e_norm);
  proj_kernel<<<BT / 16, 128, 0, stream>>>(x, proj, h_bf, h_norm);
  dim3 g(BT / MTILE, NCHUNK);
  sims_topk <<<g, 256, SMEM_TOTAL, stream>>>(emb_bf, e_norm, h_bf, h_norm, pvals, pidx);
  merge_out <<<BT, 256, 0, stream>>>(pvals, pidx, emb, out);
}
